// LightningAttention_24395414241306
// MI455X (gfx1250) — compile-verified, matrix operand order checked by reading, not run on hardware
//
#include <hip/hip_runtime.h>
#include <stdint.h>

// Lightning attention for MI455X (gfx1250, wave32, WMMA bf16 16x16x32).
// B=2, N=4096, D=2048, H=16, d=128, CHUNK=128.

#define BATCH  2
#define SEQ    4096
#define DMODEL 2048
#define NHEAD  16
#define HDIM   128
#define CCHUNK 128
#define NCHUNK (SEQ / CCHUNK)
#define ROWS   (BATCH * SEQ)      // 8192
#define QKV_N  (3 * DMODEL)       // 6144

typedef __bf16 bf16_t;
typedef __attribute__((ext_vector_type(16))) __bf16 v16bf;
typedef __attribute__((ext_vector_type(8)))  __bf16 v8bf;
typedef __attribute__((ext_vector_type(4)))  __bf16 v4bf;
typedef __attribute__((ext_vector_type(8)))  float  v8f;
typedef __attribute__((ext_vector_type(4)))  float  v4f;

__device__ __forceinline__ v8f wmma_bf16(v16bf a, v16bf b, v8f c) {
  // D = A(16x32) * B(32x16) + C, f32 accumulate
  return __builtin_amdgcn_wmma_f32_16x16x32_bf16(false, a, false, b,
                                                 (short)0, c, false, false);
}

// Async copy of 16 bytes per lane from global (saddr base + per-lane byte
// offset) into LDS (per-lane byte offset).  Tracked with ASYNCcnt; the
// imm offset is added to BOTH the global and LDS addresses.
__device__ __forceinline__ void async_copy16(unsigned lds_off, unsigned glb_off,
                                             const void* sbase) {
  asm volatile("global_load_async_to_lds_b128 %0, %1, %2"
               :: "v"(lds_off), "v"(glb_off), "s"(sbase) : "memory");
}
__device__ __forceinline__ void async_copy16_o16(unsigned lds_off, unsigned glb_off,
                                                 const void* sbase) {
  asm volatile("global_load_async_to_lds_b128 %0, %1, %2 offset:16"
               :: "v"(lds_off), "v"(glb_off), "s"(sbase) : "memory");
}
__device__ __forceinline__ void wait_async0() {
  asm volatile("s_wait_asynccnt 0" ::: "memory");
}
__device__ __forceinline__ unsigned lds_off32(const void* p) {
  return (unsigned)(uintptr_t)p;   // low 32 bits of generic ptr = LDS offset
}

// Load one WMMA operand fragment from a row-major bf16 matrix (A-style:
// lane L<16 holds row L, K = {0..7, 16..23}; lane L+16 holds row L,
// K = {8..15, 24..31}).  Works for the B operand too when the source is
// stored as N x K (i.e. B-transposed) row-major.
__device__ __forceinline__ v16bf load_frag(const bf16_t* base, int ld) {
  const int lane = threadIdx.x & 31;
  const bf16_t* p = base + (lane & 15) * ld + ((lane >> 4) << 3);
  v16bf f;
  *reinterpret_cast<v8bf*>(&f)       = *reinterpret_cast<const v8bf*>(p);
  *(reinterpret_cast<v8bf*>(&f) + 1) = *reinterpret_cast<const v8bf*>(p + 16);
  return f;
}

// Scale all 16 bf16 elements of a fragment by a (per-lane uniform) float.
__device__ __forceinline__ v16bf scale_frag(v16bf f, float s) {
  v16bf r;
#pragma unroll
  for (int i = 0; i < 16; ++i) r[i] = (bf16_t)((float)f[i] * s);
  return r;
}

// ---------------------------------------------------------------------------
// Kernel 1: QKV projection.  C(8192 x 6144) = X(8192x2048) * Wqkv^T + b.
// Writes bf16 q/k/v in (B,H,N,d) layout.
// ---------------------------------------------------------------------------
#define GLDT 72   // padded LDS row stride (elements); 144B = multiple of 16B

__global__ __launch_bounds__(256) void qkv_gemm_kernel(
    const float* __restrict__ x, const float* __restrict__ w,
    const float* __restrict__ bias,
    bf16_t* __restrict__ qw, bf16_t* __restrict__ kw, bf16_t* __restrict__ vw) {
  __shared__ bf16_t lA[128 * GLDT];
  __shared__ bf16_t lB[128 * GLDT];
  const int n0 = blockIdx.x * 128;
  const int m0 = blockIdx.y * 128;
  const int tid = threadIdx.x;
  const int wave = tid >> 5, lane = tid & 31;
  const int wr = wave >> 1;          // tile-row group (2 tile rows each)
  const int wc = wave & 1;           // tile-col group (4 tile cols each)
  const int ln = lane & 15, lm = (lane >> 4) << 3;

  v8f acc[2][4] = {};

  for (int k0 = 0; k0 < DMODEL; k0 += 64) {
    // Stage A and B tiles (128 x 64), converting f32 -> bf16.
    const int r  = tid >> 4;        // 0..15
    const int cq = tid & 15;        // float4 column index
#pragma unroll
    for (int p = 0; p < 8; ++p) {
      const int row = r + p * 16;
      v4f a4 = *reinterpret_cast<const v4f*>(x + (size_t)(m0 + row) * DMODEL + k0 + cq * 4);
      v4bf ab; ab[0]=(bf16_t)a4[0]; ab[1]=(bf16_t)a4[1]; ab[2]=(bf16_t)a4[2]; ab[3]=(bf16_t)a4[3];
      *reinterpret_cast<v4bf*>(&lA[row * GLDT + cq * 4]) = ab;
      v4f b4 = *reinterpret_cast<const v4f*>(w + (size_t)(n0 + row) * DMODEL + k0 + cq * 4);
      v4bf bb; bb[0]=(bf16_t)b4[0]; bb[1]=(bf16_t)b4[1]; bb[2]=(bf16_t)b4[2]; bb[3]=(bf16_t)b4[3];
      *reinterpret_cast<v4bf*>(&lB[row * GLDT + cq * 4]) = bb;
    }
    __syncthreads();
#pragma unroll
    for (int ks = 0; ks < 2; ++ks) {
      v16bf af[2], bfr[4];
#pragma unroll
      for (int i = 0; i < 2; ++i) af[i]  = load_frag(&lA[((wr*2+i)*16)*GLDT + ks*32], GLDT);
#pragma unroll
      for (int j = 0; j < 4; ++j) bfr[j] = load_frag(&lB[((wc*4+j)*16)*GLDT + ks*32], GLDT);
#pragma unroll
      for (int i = 0; i < 2; ++i)
#pragma unroll
        for (int j = 0; j < 4; ++j) acc[i][j] = wmma_bf16(af[i], bfr[j], acc[i][j]);
    }
    __syncthreads();
  }

  // Epilogue: add bias, convert to bf16, scatter into q/k/v (B,H,N,d).
#pragma unroll
  for (int i = 0; i < 2; ++i)
#pragma unroll
    for (int j = 0; j < 4; ++j) {
      const int col = n0 + (wc*4+j)*16 + ln;       // 0..6143
      const float bs = bias[col];
      int jj = col;
      bf16_t* dst = qw;
      if (jj >= 4096)      { dst = vw; jj -= 4096; }
      else if (jj >= 2048) { dst = kw; jj -= 2048; }
      const int h = jj >> 7, e = jj & 127;
#pragma unroll
      for (int t = 0; t < 8; ++t) {
        const int m = m0 + (wr*2+i)*16 + lm + t;
        const int b = m >> 12, n = m & 4095;
        dst[((((size_t)b * NHEAD + h) * SEQ + n) << 7) + e] = (bf16_t)(acc[i][j][t] + bs);
      }
    }
}

// ---------------------------------------------------------------------------
// Kernel 2: chunked lightning-attention recurrence.  One block per (b,h),
// 1024 threads = 32 waves.  State S (128x128 f32) lives in WMMA accumulators
// (2 tiles per wave), with a bf16 transposed LDS mirror used as the B
// operand of the inter-chunk matmul.  q staging uses async global->LDS.
// ---------------------------------------------------------------------------
#define ALD 136   // padded LDS stride (272B, multiple of 16B)

__global__ __launch_bounds__(1024) void lightning_attn_kernel(
    const bf16_t* __restrict__ qg, const bf16_t* __restrict__ kg,
    const bf16_t* __restrict__ vg, bf16_t* __restrict__ attn) {
  extern __shared__ bf16_t lds[];
  bf16_t* sq  = lds;                 // q chunk,   row-major [c][e]
  bf16_t* sk  = sq  + 128 * ALD;     // k chunk,   row-major [c][e]
  bf16_t* skT = sk  + 128 * ALD;     // (k*decay_k)^T      [e][c]
  bf16_t* svT = skT + 128 * ALD;     // v^T                [e'][c]
  bf16_t* sS  = svT + 128 * ALD;     // S^T bf16           [e'][e]
  bf16_t* ssc = sS  + 128 * ALD;     // masked scores      [c][c']

  const int bh = blockIdx.x;
  const int b = bh >> 4, h = bh & 15;
  const int tid = threadIdx.x;
  const int wave = tid >> 5, lane = tid & 31;
  const int ln = lane & 15, lm = (lane >> 4) << 3;
  const int tm = wave >> 2;              // output/state tile row  (0..7)
  const int tn0 = (wave & 3) << 1;       // output/state tile cols (tn0, tn0+1)

  const float slope  = exp2f(-0.5f * (float)(h + 1));   // ALiBi slope
  const float l2lam  = -slope * 1.44269504088896340736f; // log2(exp(-slope))
  const float lam_c  = exp2f(l2lam * (float)CCHUNK);

  const size_t base = ((size_t)b * NHEAD + h) * SEQ * HDIM;

  v8f S0 = {}, S1 = {};   // this wave's two 16x16 tiles of S

  for (int i = tid; i < 128 * ALD; i += 1024) sS[i] = (bf16_t)0.0f;
  __syncthreads();

  for (int ch = 0; ch < NCHUNK; ++ch) {
    const size_t gofs = base + (size_t)ch * CCHUNK * HDIM;
    // ---- stage chunk: q (async), k, k^T (decayed), v^T ----
    {
      const int r  = tid >> 3;           // 0..127 (chunk position c)
      const int c0 = (tid & 7) << 4;     // feature start (16 elements)

      // q: straight bf16 copy -> async global->LDS (32B per thread)
      const unsigned lq = lds_off32(sq + r * ALD + c0);
      const unsigned gq = (unsigned)((r * HDIM + c0) * sizeof(bf16_t));
      const bf16_t* qbase = qg + gofs;   // uniform (SGPR) base
      async_copy16(lq, gq, qbase);
      async_copy16_o16(lq, gq, qbase);

      const float dk = exp2f(l2lam * (float)(CCHUNK - 1 - r));  // lam^(127-c)
      v8bf k0 = *reinterpret_cast<const v8bf*>(kg + gofs + (size_t)r * HDIM + c0);
      v8bf k1 = *reinterpret_cast<const v8bf*>(kg + gofs + (size_t)r * HDIM + c0 + 8);
      *reinterpret_cast<v8bf*>(sk + r * ALD + c0)     = k0;
      *reinterpret_cast<v8bf*>(sk + r * ALD + c0 + 8) = k1;
#pragma unroll
      for (int t = 0; t < 8; ++t) {
        skT[(c0 + t)     * ALD + r] = (bf16_t)((float)k0[t] * dk);
        skT[(c0 + 8 + t) * ALD + r] = (bf16_t)((float)k1[t] * dk);
      }
      v8bf v0 = *reinterpret_cast<const v8bf*>(vg + gofs + (size_t)r * HDIM + c0);
      v8bf v1 = *reinterpret_cast<const v8bf*>(vg + gofs + (size_t)r * HDIM + c0 + 8);
#pragma unroll
      for (int t = 0; t < 8; ++t) {
        svT[(c0 + t)     * ALD + r] = v0[t];
        svT[(c0 + 8 + t) * ALD + r] = v1[t];
      }
      wait_async0();   // q async copies complete before the barrier
    }
    __syncthreads();

    // ---- Phase A: scores = (q @ k^T) * decay mask ----
    {
      v8f a0 = {}, a1 = {};
#pragma unroll
      for (int ks = 0; ks < 4; ++ks) {
        v16bf qa  = load_frag(sq + (tm   * 16) * ALD + ks * 32, ALD);
        v16bf kb0 = load_frag(sk + (tn0  * 16) * ALD + ks * 32, ALD);
        v16bf kb1 = load_frag(sk + ((tn0 + 1) * 16) * ALD + ks * 32, ALD);
        a0 = wmma_bf16(qa, kb0, a0);
        a1 = wmma_bf16(qa, kb1, a1);
      }
#pragma unroll
      for (int t = 0; t < 8; ++t) {
        const int c   = tm * 16 + lm + t;
        const int cn0 = tn0 * 16 + ln;
        const int cn1 = cn0 + 16;
        const float m0 = (c >= cn0) ? exp2f(l2lam * (float)(c - cn0)) : 0.0f;
        const float m1 = (c >= cn1) ? exp2f(l2lam * (float)(c - cn1)) : 0.0f;
        ssc[c * ALD + cn0] = (bf16_t)(a0[t] * m0);
        ssc[c * ALD + cn1] = (bf16_t)(a1[t] * m1);
      }
    }
    __syncthreads();

    // ---- Phase B: out = scores @ v  +  (q * lam^(c+1)) @ S ----
    {
      v8f o0 = {}, o1 = {};
      const float dq = exp2f(l2lam * (float)(tm * 16 + ln + 1));  // this lane's A row
#pragma unroll
      for (int ks = 0; ks < 4; ++ks) {
        v16bf sa  = load_frag(ssc + (tm * 16) * ALD + ks * 32, ALD);
        v16bf vb0 = load_frag(svT + (tn0 * 16) * ALD + ks * 32, ALD);
        v16bf vb1 = load_frag(svT + ((tn0 + 1) * 16) * ALD + ks * 32, ALD);
        o0 = wmma_bf16(sa, vb0, o0);
        o1 = wmma_bf16(sa, vb1, o1);
        v16bf qa  = scale_frag(load_frag(sq + (tm * 16) * ALD + ks * 32, ALD), dq);
        v16bf sb0 = load_frag(sS + (tn0 * 16) * ALD + ks * 32, ALD);
        v16bf sb1 = load_frag(sS + ((tn0 + 1) * 16) * ALD + ks * 32, ALD);
        o0 = wmma_bf16(qa, sb0, o0);
        o1 = wmma_bf16(qa, sb1, o1);
      }
#pragma unroll
      for (int t = 0; t < 8; ++t) {
        const int c = tm * 16 + lm + t;
        const int n = ch * CCHUNK + c;
        const size_t rb = ((size_t)b * SEQ + n) * DMODEL + h * HDIM;
        attn[rb + tn0 * 16 + ln]      = (bf16_t)o0[t];
        attn[rb + tn0 * 16 + 16 + ln] = (bf16_t)o1[t];
      }
    }

    // ---- Phase C: S = lam^C * S + (k*decay_k)^T @ v  (in accumulators) ----
    {
#pragma unroll
      for (int t = 0; t < 8; ++t) { S0[t] *= lam_c; S1[t] *= lam_c; }
#pragma unroll
      for (int ks = 0; ks < 4; ++ks) {
        v16bf ka  = load_frag(skT + (tm * 16) * ALD + ks * 32, ALD);
        v16bf vb0 = load_frag(svT + (tn0 * 16) * ALD + ks * 32, ALD);
        v16bf vb1 = load_frag(svT + ((tn0 + 1) * 16) * ALD + ks * 32, ALD);
        S0 = wmma_bf16(ka, vb0, S0);
        S1 = wmma_bf16(ka, vb1, S1);
      }
    }
    __syncthreads();           // everyone done reading old sS
    // publish S^T (bf16) for next chunk's inter matmul
#pragma unroll
    for (int t = 0; t < 8; ++t) {
      const int e  = tm * 16 + lm + t;
      const int e0 = tn0 * 16 + ln, e1 = e0 + 16;
      sS[e0 * ALD + e] = (bf16_t)S0[t];
      sS[e1 * ALD + e] = (bf16_t)S1[t];
    }
    __syncthreads();
  }
}

// ---------------------------------------------------------------------------
// Kernel 3: output projection.  out(8192x2048) = attn_bf @ Wout^T + b (f32).
// A tile (already bf16) staged with async global->LDS copies.
// ---------------------------------------------------------------------------
__global__ __launch_bounds__(256) void out_gemm_kernel(
    const bf16_t* __restrict__ a, const float* __restrict__ w,
    const float* __restrict__ bias, float* __restrict__ out) {
  __shared__ bf16_t lA[128 * GLDT];
  __shared__ bf16_t lB[128 * GLDT];
  const int n0 = blockIdx.x * 128;
  const int m0 = blockIdx.y * 128;
  const int tid = threadIdx.x;
  const int wave = tid >> 5, lane = tid & 31;
  const int wr = wave >> 1, wc = wave & 1;
  const int ln = lane & 15, lm = (lane >> 4) << 3;

  const bf16_t* abase = a + (size_t)m0 * DMODEL;  // uniform (SGPR) base

  v8f acc[2][4] = {};

  for (int k0 = 0; k0 < DMODEL; k0 += 64) {
    {
      const int r  = tid >> 3;         // 0..31
      const int c8 = (tid & 7) << 3;   // 8-element column start
#pragma unroll
      for (int p = 0; p < 4; ++p) {
        const int row = r + p * 32;
        const unsigned lofs = lds_off32(&lA[row * GLDT + c8]);
        const unsigned gofs = (unsigned)((row * DMODEL + k0 + c8) * sizeof(bf16_t));
        async_copy16(lofs, gofs, abase);
      }
      const int r2 = tid >> 4, cq = tid & 15;
#pragma unroll
      for (int p = 0; p < 8; ++p) {
        const int row = r2 + p * 16;
        v4f b4 = *reinterpret_cast<const v4f*>(w + (size_t)(n0 + row) * DMODEL + k0 + cq * 4);
        v4bf bb; bb[0]=(bf16_t)b4[0]; bb[1]=(bf16_t)b4[1]; bb[2]=(bf16_t)b4[2]; bb[3]=(bf16_t)b4[3];
        *reinterpret_cast<v4bf*>(&lB[row * GLDT + cq * 4]) = bb;
      }
      wait_async0();
    }
    __syncthreads();
#pragma unroll
    for (int ks = 0; ks < 2; ++ks) {
      v16bf af[2], bfr[4];
#pragma unroll
      for (int i = 0; i < 2; ++i) af[i]  = load_frag(&lA[((wr*2+i)*16)*GLDT + ks*32], GLDT);
#pragma unroll
      for (int j = 0; j < 4; ++j) bfr[j] = load_frag(&lB[((wc*4+j)*16)*GLDT + ks*32], GLDT);
#pragma unroll
      for (int i = 0; i < 2; ++i)
#pragma unroll
        for (int j = 0; j < 4; ++j) acc[i][j] = wmma_bf16(af[i], bfr[j], acc[i][j]);
    }
    __syncthreads();
  }

#pragma unroll
  for (int i = 0; i < 2; ++i)
#pragma unroll
    for (int j = 0; j < 4; ++j) {
      const int col = n0 + (wc*4+j)*16 + ln;
      const float bs = bias[col];
#pragma unroll
      for (int t = 0; t < 8; ++t) {
        const int m = m0 + (wr*2+i)*16 + lm + t;
        out[(size_t)m * DMODEL + col] = acc[i][j][t] + bs;
      }
    }
}

// ---------------------------------------------------------------------------
extern "C" void kernel_launch(void* const* d_in, const int* in_sizes, int n_in,
                              void* d_out, int out_size, void* d_ws, size_t ws_size,
                              hipStream_t stream) {
  (void)in_sizes; (void)n_in; (void)out_size; (void)ws_size;
  const float* x     = (const float*)d_in[0];
  const float* w_qkv = (const float*)d_in[1];
  const float* b_qkv = (const float*)d_in[2];
  const float* w_out = (const float*)d_in[3];
  const float* b_out = (const float*)d_in[4];
  float* out = (float*)d_out;

  // Workspace layout (bf16): q | k | v  in (B,H,N,d), then attn in (B,N,D).
  // Total = 4 * 16,777,216 * 2 = 128 MiB.
  bf16_t* ws = (bf16_t*)d_ws;
  const size_t per = (size_t)BATCH * NHEAD * SEQ * HDIM;
  bf16_t* qw   = ws;
  bf16_t* kw   = qw + per;
  bf16_t* vw   = kw + per;
  bf16_t* attn = vw + per;

  qkv_gemm_kernel<<<dim3(QKV_N / 128, ROWS / 128), 256, 0, stream>>>(
      x, w_qkv, b_qkv, qw, kw, vw);

  const size_t attn_lds = (size_t)6 * 128 * ALD * sizeof(bf16_t);  // ~204 KB
  hipFuncSetAttribute((const void*)lightning_attn_kernel,
                      hipFuncAttributeMaxDynamicSharedMemorySize, (int)attn_lds);
  lightning_attn_kernel<<<BATCH * NHEAD, 1024, attn_lds, stream>>>(qw, kw, vw, attn);

  out_gemm_kernel<<<dim3(DMODEL / 128, ROWS / 128), 256, 0, stream>>>(
      attn, w_out, b_out, out);
}
